// CustomSimpleLSTM_54915451847143
// MI455X (gfx1250) — matrix operand order repaired and checked by reading, not compiled, not run on hardware
//
#include <hip/hip_runtime.h>
#include <hip/hip_bf16.h>
#include <math.h>

// ---------------- WMMA types (CDNA5 / gfx1250, wave32) ----------------
typedef __attribute__((ext_vector_type(16))) __bf16 v16bf;
typedef __attribute__((ext_vector_type(8)))  float  v8f;

// round-to-nearest-even f32 -> bf16
__device__ __forceinline__ unsigned short f32_to_bf16(float f) {
    union { float f; unsigned u; } x; x.f = f;
    unsigned u = x.u;
    u += 0x7FFFu + ((u >> 16) & 1u);
    return (unsigned short)(u >> 16);
}

// CDNA5 has a hardware TANH transcendental (V_TANH_F32); use it if the
// toolchain exposes the builtin, else fall back to libm.
__device__ __forceinline__ float hw_tanh(float x) {
#if __has_builtin(__builtin_amdgcn_tanhf)
    return __builtin_amdgcn_tanhf(x);
#else
    return tanhf(x);
#endif
}

// sigmoid(x) = 0.5*tanh(x/2) + 0.5  -> one trans op + one FMA
__device__ __forceinline__ float fast_sigmoid(float x) {
    return __builtin_fmaf(0.5f, hw_tanh(0.5f * x), 0.5f);
}

// ---------------- prep kernels ----------------
// W: [K,N] f32 row-major  ->  WT: [N,K] bf16 row-major (so B-fragments are contiguous in K)
__global__ void k_transpose_bf16(const float* __restrict__ W, unsigned short* __restrict__ WT,
                                 int K, int N) {
    int idx = blockIdx.x * blockDim.x + threadIdx.x;
    if (idx >= K * N) return;
    int n = idx / K;
    int k = idx - n * K;
    WT[idx] = f32_to_bf16(W[(long)k * N + n]);
}

__global__ void k_to_bf16(const float* __restrict__ src, unsigned short* __restrict__ dst, int n) {
    int i = blockIdx.x * blockDim.x + threadIdx.x;
    if (i < n) dst[i] = f32_to_bf16(src[i]);
}

// ---------------- fused LSTM step (one layer, one timestep) ----------------
// gates[64, 4*512] = cat(A0|A1)[64, K] @ W[K, 2048] + bias, then cell update.
// Grid: 128 blocks x 32 threads (1 wave per block). Block b -> tile
//   m0 = (b/32)*16 (batch rows), n0 = (b%32)*16 (hidden cols).
// The wave accumulates gate g in acc[g]; all four gates share identical
// (m,n) lane coordinates in the f32 C-layout, so the nonlinear cell update
// is done fully in registers.
template <int KSTEPS>
__global__ __launch_bounds__(32)
void k_lstm_step(const unsigned short* __restrict__ A0, int a0_stride, int k_split,
                 const unsigned short* __restrict__ A1, int a1_stride,
                 const unsigned short* __restrict__ WT, int K,
                 const float* __restrict__ bias,
                 float* __restrict__ c_state,
                 unsigned short* __restrict__ h_out_bf,
                 float* __restrict__ h_out_f32) {
    const int lane = threadIdx.x;          // 0..31
    const int half = lane >> 4;            // 0 or 1
    const int lrow = lane & 15;            // 0..15
    const int m0 = (blockIdx.x >> 5) * 16; // 4 M-tiles
    const int n0 = (blockIdx.x & 31) * 16; // 32 N-tiles (per gate)

    v8f acc[4] = {};                       // i, f, g, o accumulators (f32)

    #pragma unroll 2
    for (int ks = 0; ks < KSTEPS; ++ks) {
        const int k0 = ks * 32;
        // ---- A fragment: 16x32 bf16. ISA layout: lane<16 holds M=lane,
        // K = k0+0..7 and k0+16..23; lane>=16 holds M=lane-16, K = +8..15, +24..31.
        const unsigned short* arow;
        int kk;
        if (k0 < k_split) { arow = A0 + (long)(m0 + lrow) * a0_stride; kk = k0; }
        else             { arow = A1 + (long)(m0 + lrow) * a1_stride; kk = k0 - k_split; }
        union { uint4 q[2]; v16bf v; } afrag;
        afrag.q[0] = *(const uint4*)(arow + kk + half * 8);
        afrag.q[1] = *(const uint4*)(arow + kk + 16 + half * 8);

        // ---- B fragments per gate: 32x16 bf16. Lane n (0..15): col n, K=k0+0..7, k0+16..23;
        // lane n+16: col n, K=k0+8..15, k0+24..31 -> two 16B loads per lane from WT[col][*].
        #pragma unroll
        for (int g = 0; g < 4; ++g) {
            const int col = g * 512 + n0 + lrow;
            const unsigned short* bptr = WT + (long)col * K + k0 + half * 8;
            union { uint4 q[2]; v16bf v; } bfrag;
            bfrag.q[0] = *(const uint4*)(bptr);
            bfrag.q[1] = *(const uint4*)(bptr + 16);
            acc[g] = __builtin_amdgcn_wmma_f32_16x16x32_bf16(
                false, afrag.v, false, bfrag.v, (short)0, acc[g], false, false);
        }
    }

    // ---- epilogue: bias + LSTM cell, all in registers.
    // f32 C/D layout: vgpr r, lane l -> m = m0 + r + 8*(l>=16), n = n0 + (l&15).
    const int n = n0 + lrow;
    const float bi = bias[          n];
    const float bf_= bias[ 512 +    n];
    const float bg = bias[1024 +    n];
    const float bo = bias[1536 +    n];
    #pragma unroll
    for (int r = 0; r < 8; ++r) {
        const int  m   = m0 + r + half * 8;
        const long idx = (long)m * 512 + n;
        const float iv = fast_sigmoid(acc[0][r] + bi);
        const float fv = fast_sigmoid(acc[1][r] + bf_);
        const float gv = hw_tanh     (acc[2][r] + bg);
        const float ov = fast_sigmoid(acc[3][r] + bo);
        const float cn = fv * c_state[idx] + iv * gv;
        c_state[idx] = cn;
        const float h = ov * hw_tanh(cn);
        h_out_bf[idx] = f32_to_bf16(h);
        if (h_out_f32) h_out_f32[idx] = h;
    }
}

// ---------------- output head: pred[m] = h_top[m,:] . Wfc + bfc ----------------
__global__ __launch_bounds__(32)
void k_fc(const float* __restrict__ h, const float* __restrict__ Wfc,
          const float* __restrict__ bfc, float* __restrict__ out) {
    int m = blockIdx.x * blockDim.x + threadIdx.x;
    if (m >= 64) return;
    const float* row = h + (long)m * 512;
    float s = 0.f;
    #pragma unroll 4
    for (int k = 0; k < 512; ++k) s += row[k] * Wfc[k];
    out[m] = s + bfc[0];
}

// ---------------- host orchestration ----------------
extern "C" void kernel_launch(void* const* d_in, const int* in_sizes, int n_in,
                              void* d_out, int out_size, void* d_ws, size_t ws_size,
                              hipStream_t stream) {
    (void)in_sizes; (void)n_in; (void)out_size; (void)ws_size;
    const float* x   = (const float*)d_in[0]; // [64,512,64]
    const float* W0  = (const float*)d_in[1]; // [576,2048]
    const float* b0  = (const float*)d_in[2]; // [2048]
    const float* W1  = (const float*)d_in[3]; // [1024,2048]
    const float* b1  = (const float*)d_in[4]; // [2048]
    const float* Wfc = (const float*)d_in[5]; // [512,1]
    const float* bfc = (const float*)d_in[6]; // [1]
    float* out = (float*)d_out;               // [64]

    constexpr int B = 64, S = 512, I = 64, H = 512;
    constexpr int K0 = I + H;   // 576
    constexpr int K1 = H + H;   // 1024
    constexpr int NG = 4 * H;   // 2048

    char* p = (char*)d_ws;
    auto carve = [&](size_t bytes) -> void* {
        void* r = (void*)p; p += (bytes + 255) & ~(size_t)255; return r;
    };
    unsigned short* W0T  = (unsigned short*)carve((size_t)NG * K0 * 2); // [2048][576]
    unsigned short* W1T  = (unsigned short*)carve((size_t)NG * K1 * 2); // [2048][1024]
    unsigned short* xbf  = (unsigned short*)carve((size_t)B * S * I * 2);
    unsigned short* h0[2], *h1[2];
    h0[0] = (unsigned short*)carve((size_t)B * H * 2);
    h0[1] = (unsigned short*)carve((size_t)B * H * 2);
    h1[0] = (unsigned short*)carve((size_t)B * H * 2);
    h1[1] = (unsigned short*)carve((size_t)B * H * 2);
    float* c0  = (float*)carve((size_t)B * H * 4);
    float* c1  = (float*)carve((size_t)B * H * 4);
    float* h1f = (float*)carve((size_t)B * H * 4);

    // ---- prep: bf16 weight transposes, bf16 x, zero initial state ----
    {
        int t0 = K0 * NG;
        k_transpose_bf16<<<(t0 + 255) / 256, 256, 0, stream>>>(W0, W0T, K0, NG);
        int t1 = K1 * NG;
        k_transpose_bf16<<<(t1 + 255) / 256, 256, 0, stream>>>(W1, W1T, K1, NG);
        int tx = B * S * I;
        k_to_bf16<<<(tx + 255) / 256, 256, 0, stream>>>(x, xbf, tx);
        hipMemsetAsync(h0[0], 0, (size_t)B * H * 2, stream);
        hipMemsetAsync(h1[0], 0, (size_t)B * H * 2, stream);
        hipMemsetAsync(c0,    0, (size_t)B * H * 4, stream);
        hipMemsetAsync(c1,    0, (size_t)B * H * 4, stream);
    }

    // ---- recurrence: 2 fused-layer step launches per timestep ----
    for (int t = 0; t < S; ++t) {
        const int cur = t & 1, nxt = (t + 1) & 1;
        // layer 0: A = cat(x_t [64,64], h0_prev [64,512]); K = 576, 18 k-steps
        k_lstm_step<K0 / 32><<<128, 32, 0, stream>>>(
            xbf + (long)t * I, S * I, I,
            h0[cur], H,
            W0T, K0, b0, c0, h0[nxt], nullptr);
        // layer 1: A = cat(h0_next [64,512], h1_prev [64,512]); K = 1024, 32 k-steps
        k_lstm_step<K1 / 32><<<128, 32, 0, stream>>>(
            h0[nxt], H, H,
            h1[cur], H,
            W1T, K1, b1, c1, h1[nxt], h1f);
    }

    // ---- head on final top-layer hidden state ----
    k_fc<<<2, 32, 0, stream>>>(h1f, Wfc, bfc, out);
}
